// RelAwareRelFeature_69286412419099
// MI455X (gfx1250) — matrix operand order repaired and checked by reading, not compiled, not run on hardware
//
#include <hip/hip_runtime.h>
#include <hip/hip_bf16.h>

typedef __attribute__((ext_vector_type(16))) __bf16 v16bf;
typedef __attribute__((ext_vector_type(8)))  __bf16 bf16x8;
typedef __attribute__((ext_vector_type(8)))  float  v8f;
typedef __attribute__((ext_vector_type(4)))  float  v4f;

union Frag16 { v16bf v; bf16x8 h[2]; };

static __device__ __forceinline__ v8f wmma_bf16(v16bf a, v16bf b, v8f c) {
    return __builtin_amdgcn_wmma_f32_16x16x32_bf16(false, a, false, b, (short)0, c, false, false);
}

#define BB 4
#define NN 64
#define PP 4032
#define DIN 4096
#define HH 512
#define EE 768
#define GG 128
#define COBJ 151
#define RR 50
#define MTOT (BB * PP)          // 16128

// ---------------------------------------------------------------- utilities
__global__ __launch_bounds__(256) void k_zero(float* p, int n) {
    for (int i = blockIdx.x * blockDim.x + threadIdx.x; i < n; i += gridDim.x * blockDim.x)
        p[i] = 0.f;
}

// W[K][N] f32 -> Wt[N][K] bf16
__global__ __launch_bounds__(256) void k_cvt_wt(const float* __restrict__ W,
                                                __bf16* __restrict__ Wt, int K, int N) {
    int total = K * N;
    for (int idx = blockIdx.x * blockDim.x + threadIdx.x; idx < total;
         idx += gridDim.x * blockDim.x) {
        int n = idx / K;
        int k = idx - n * K;
        Wt[idx] = (__bf16)W[(size_t)k * N + n];
    }
}

// ---------------------------------------------------------------- pos / sem
__global__ __launch_bounds__(256) void k_prep(const float* __restrict__ box,
                                              const float* __restrict__ plog,
                                              const float* __restrict__ emb,
                                              const float* __restrict__ pw1,
                                              const float* __restrict__ pb1,
                                              const float* __restrict__ pw2,
                                              const float* __restrict__ pb2,
                                              float* __restrict__ pos,
                                              float* __restrict__ sem) {
    int bn = blockIdx.x;           // 0..255 (b*64+n)
    int t  = threadIdx.x;
    __shared__ float tmp[256];
    __shared__ float w151[COBJ];
    __shared__ float hbuf[GG];
    __shared__ float sbc[2];

    const float* lg = plog + (size_t)bn * COBJ;
    float x = (t < COBJ) ? lg[t] : -1e30f;
    tmp[t] = x;
    __syncthreads();
    if (t == 0) { float m = -1e30f; for (int i = 0; i < COBJ; ++i) m = fmaxf(m, tmp[i]); sbc[0] = m; }
    __syncthreads();
    float e = (t < COBJ) ? __expf(x - sbc[0]) : 0.f;
    tmp[t] = e;
    __syncthreads();
    if (t == 0) { float s = 0.f; for (int i = 0; i < COBJ; ++i) s += tmp[i]; sbc[1] = s; }
    __syncthreads();
    if (t < COBJ) w151[t] = e / sbc[1];

    const float* bx = box + (size_t)bn * 9;
    if (t < GG) {
        float h = pb1[t];
        for (int k = 0; k < 9; ++k) h += bx[k] * pw1[k * GG + t];
        hbuf[t] = fmaxf(h, 0.f);
    }
    __syncthreads();
    if (t < GG) {
        float o = pb2[t];
        for (int k = 0; k < GG; ++k) o += hbuf[k] * pw2[k * GG + t];
        pos[(size_t)bn * GG + t] = o;
    }
    for (int e0 = t; e0 < EE; e0 += 256) {
        float a = 0.f;
        for (int c = 0; c < COBJ; ++c) a += w151[c] * emb[(size_t)c * EE + e0];
        sem[(size_t)bn * EE + e0] = a;
    }
}

// ---------------------------------------------------------------- relu(pair) bf16
__global__ __launch_bounds__(256) void k_pair(const int* __restrict__ pidx,
                                              const float* __restrict__ pos,
                                              const float* __restrict__ sem,
                                              __bf16* __restrict__ pair) {
    int bp = blockIdx.x;                 // b*4032+p
    int t  = threadIdx.x;
    int b  = bp / PP;
    int i  = pidx[(size_t)bp * 2];
    int j  = pidx[(size_t)bp * 2 + 1];
    const float* pi = pos + ((size_t)b * NN + i) * GG;
    const float* si = sem + ((size_t)b * NN + i) * EE;
    const float* pj = pos + ((size_t)b * NN + j) * GG;
    const float* sj = sem + ((size_t)b * NN + j) * EE;
    __bf16* row = pair + (size_t)bp * 1792;
    for (int idx = t; idx < 1792; idx += 256) {
        float v;
        if (idx < 128)       v = pi[idx];
        else if (idx < 896)  v = si[idx - 128];
        else if (idx < 1024) v = pj[idx - 896];
        else                 v = sj[idx - 1024];
        row[idx] = (__bf16)fmaxf(v, 0.f);
    }
}

// ---------------------------------------------------------------- WMMA GEMM
// C[M,N] = op(A)[M,K] * B[K,N] + bias;  Bt given row-major [N][K] bf16.
// Block covers 32 rows x full N=512 (A streamed from HBM exactly once;
// B panel stays L2-resident).  8 waves = 2(M) x 4(N), wave tile 16x128.
// AMODE 0: A bf16 row-major.  AMODE 1: A f32 row-major, relu applied.
// STORET false: out bf16 [M][512].  STORET true: Vt layout [b][col][p] bf16.
template <int AMODE, bool STORET>
__global__ __launch_bounds__(256) void gemm_bf16(const void* __restrict__ Ap, int lda,
                                                 const __bf16* __restrict__ Bt, int ldb,
                                                 const float* __restrict__ bias,
                                                 void* __restrict__ outp, int K) {
    const int tid  = threadIdx.x;
    const int lane = tid & 31;
    const int w    = tid >> 5;       // 0..7
    const int wm   = w & 1;          // 2 M-waves
    const int wn   = w >> 1;         // 4 N-waves
    const int half = lane >> 4;
    const int l16  = lane & 15;
    const int m0   = blockIdx.x * 32 + wm * 16;
    const int n0   = wn * 128;

    const float*  Af = (const float*)Ap;
    const __bf16* Ab = (const __bf16*)Ap;
    const int mrow = m0 + l16;

    v8f acc[8];
#pragma unroll
    for (int f = 0; f < 8; ++f) acc[f] = (v8f){0.f,0.f,0.f,0.f,0.f,0.f,0.f,0.f};

    for (int k0 = 0; k0 < K; k0 += 32) {
        Frag16 a;
        if (AMODE == 1) {
            const float* ap = Af + (size_t)mrow * lda + k0 + 8 * half;
            v4f f0 = *(const v4f*)(ap);
            v4f f1 = *(const v4f*)(ap + 4);
            v4f f2 = *(const v4f*)(ap + 16);
            v4f f3 = *(const v4f*)(ap + 20);
#pragma unroll
            for (int i = 0; i < 4; ++i) {
                a.v[i]      = (__bf16)fmaxf(f0[i], 0.f);
                a.v[4 + i]  = (__bf16)fmaxf(f1[i], 0.f);
                a.v[8 + i]  = (__bf16)fmaxf(f2[i], 0.f);
                a.v[12 + i] = (__bf16)fmaxf(f3[i], 0.f);
            }
        } else {
            const __bf16* ap = Ab + (size_t)mrow * lda + k0 + 8 * half;
            a.h[0] = *(const bf16x8*)(ap);
            a.h[1] = *(const bf16x8*)(ap + 16);
        }
#pragma unroll
        for (int f = 0; f < 8; ++f) {
            const __bf16* bp = Bt + (size_t)(n0 + f * 16 + l16) * ldb + k0 + 8 * half;
            Frag16 bb;
            bb.h[0] = *(const bf16x8*)(bp);
            bb.h[1] = *(const bf16x8*)(bp + 16);
            acc[f] = wmma_bf16(a.v, bb.v, acc[f]);
        }
    }

    if (!STORET) {
        __bf16* out = (__bf16*)outp;
#pragma unroll
        for (int f = 0; f < 8; ++f) {
            int   col = n0 + f * 16 + l16;
            float bv  = bias[col];
#pragma unroll
            for (int v = 0; v < 8; ++v) {
                int row = m0 + v + 8 * half;
                out[(size_t)row * HH + col] = (__bf16)(acc[f][v] + bv);
            }
        }
    } else {
        __bf16* Vt = (__bf16*)outp;
        int b  = m0 / PP;
        int p0 = m0 - b * PP + 8 * half;
#pragma unroll
        for (int f = 0; f < 8; ++f) {
            int   col = n0 + f * 16 + l16;
            float bv  = bias[col];
            bf16x8 t;
#pragma unroll
            for (int v = 0; v < 8; ++v) t[v] = (__bf16)(acc[f][v] + bv);
            *(bf16x8*)(Vt + ((size_t)b * HH + col) * PP + p0) = t;
        }
    }
}

// ---------------------------------------------------------------- attention + head
// grid (252, B), 256 threads (8 waves).  16 query rows per block.
__global__ __launch_bounds__(256) void k_attn(const __bf16* __restrict__ Qb,
                                              const __bf16* __restrict__ Kb,
                                              const __bf16* __restrict__ Vt,
                                              const float* __restrict__ ln_g,
                                              const float* __restrict__ ln_b,
                                              const float* __restrict__ cls_w,
                                              const float* __restrict__ cls_b,
                                              const float* __restrict__ fus_w,
                                              const float* __restrict__ fus_b,
                                              const int* __restrict__ pidx,
                                              float* __restrict__ out_logits,
                                              float* __restrict__ mats) {
    const int b    = blockIdx.y;
    const int q0   = blockIdx.x * 16;
    const int tid  = threadIdx.x;
    const int lane = tid & 31;
    const int w    = tid >> 5;
    const int half = lane >> 4;
    const int l16  = lane & 15;

    __shared__ __bf16 Qs[16 * 520];     // padded row stride
    __shared__ __bf16 Ps[16 * 136];
    __shared__ float  cmax[8 * 16];
    __shared__ float  csum[8 * 16];
    __shared__ float  nm[16], sc[16], run_m[16], run_l[16];
    __shared__ float  Ob[16 * 516];
    __shared__ float  Lg[16 * 52];

    // stage Q tile (16 x 512 bf16)
    const __bf16* Qg = Qb + ((size_t)b * PP + q0) * HH;
    for (int i = tid; i < 16 * 64; i += 256) {
        int row = i >> 6, c8 = i & 63;
        *(bf16x8*)(Qs + row * 520 + c8 * 8) = *(const bf16x8*)(Qg + (size_t)row * HH + c8 * 8);
    }
    if (tid < 16) { run_m[tid] = -1e30f; run_l[tid] = 0.f; }

    v8f acc[4];
#pragma unroll
    for (int f = 0; f < 4; ++f) acc[f] = (v8f){0.f,0.f,0.f,0.f,0.f,0.f,0.f,0.f};
    __syncthreads();

    const float rs = 0.04419417382415922f;   // 1/sqrt(512)

    for (int c = 0; c < 32; ++c) {
        const int kb    = c * 128 + w * 16;
        const bool valid = (kb < PP);        // wave-uniform

        v8f s = (v8f){0.f,0.f,0.f,0.f,0.f,0.f,0.f,0.f};
        if (valid) {
            const __bf16* Kg = Kb + ((size_t)b * PP + kb) * HH;
            for (int k0 = 0; k0 < HH; k0 += 32) {
                Frag16 a, bb;
                const __bf16* qp = Qs + l16 * 520 + k0 + 8 * half;
                a.h[0] = *(const bf16x8*)(qp);
                a.h[1] = *(const bf16x8*)(qp + 16);
                const __bf16* kp = Kg + (size_t)l16 * HH + k0 + 8 * half;
                bb.h[0] = *(const bf16x8*)(kp);
                bb.h[1] = *(const bf16x8*)(kp + 16);
                s = wmma_bf16(a.v, bb.v, s);
            }
        }
        float sv[8];
#pragma unroll
        for (int v = 0; v < 8; ++v) sv[v] = valid ? s[v] * rs : -1e30f;

        // per-row max across the 16 lanes of this half
        float mx[8];
#pragma unroll
        for (int v = 0; v < 8; ++v) {
            float x = sv[v];
            for (int d = 1; d < 16; d <<= 1) x = fmaxf(x, __shfl_xor(x, d, 16));
            mx[v] = x;
        }
        if (l16 == 0) {
#pragma unroll
            for (int v = 0; v < 8; ++v) cmax[w * 16 + v + 8 * half] = mx[v];
        }
        __syncthreads();
        if (tid < 16) {
            float mo = run_m[tid], mn = mo;
            for (int ww = 0; ww < 8; ++ww) mn = fmaxf(mn, cmax[ww * 16 + tid]);
            nm[tid] = mn; sc[tid] = __expf(mo - mn); run_m[tid] = mn;
        }
        __syncthreads();

        float scl[8], nml[8];
#pragma unroll
        for (int v = 0; v < 8; ++v) { int r = v + 8 * half; scl[v] = sc[r]; nml[v] = nm[r]; }
#pragma unroll
        for (int f = 0; f < 4; ++f)
#pragma unroll
            for (int v = 0; v < 8; ++v) acc[f][v] *= scl[v];

        float p[8];
#pragma unroll
        for (int v = 0; v < 8; ++v) p[v] = __expf(sv[v] - nml[v]);
        float rsum[8];
#pragma unroll
        for (int v = 0; v < 8; ++v) {
            float x = p[v];
            for (int d = 1; d < 16; d <<= 1) x += __shfl_xor(x, d, 16);
            rsum[v] = x;
        }
        if (l16 == 0) {
#pragma unroll
            for (int v = 0; v < 8; ++v) csum[w * 16 + v + 8 * half] = rsum[v];
        }
#pragma unroll
        for (int v = 0; v < 8; ++v)
            Ps[(v + 8 * half) * 136 + w * 16 + l16] = (__bf16)p[v];
        __syncthreads();

        if (tid < 16) {
            float s2 = 0.f;
            for (int ww = 0; ww < 8; ++ww) s2 += csum[ww * 16 + tid];
            run_l[tid] = run_l[tid] * sc[tid] + s2;
        }
        // P(16x128) @ V(128x64 cols owned by this wave)
        const int kkmax = (PP - c * 128 >= 128) ? 128 : (PP - c * 128);
        const __bf16* Vg = Vt + (size_t)b * HH * PP + (size_t)c * 128;
        for (int kk = 0; kk < kkmax; kk += 32) {
            Frag16 a;
            const __bf16* pp = Ps + l16 * 136 + kk + 8 * half;
            a.h[0] = *(const bf16x8*)(pp);
            a.h[1] = *(const bf16x8*)(pp + 16);
#pragma unroll
            for (int f = 0; f < 4; ++f) {
                int col = w * 64 + f * 16 + l16;
                const __bf16* vp = Vg + (size_t)col * PP + kk + 8 * half;
                Frag16 bb;
                bb.h[0] = *(const bf16x8*)(vp);
                bb.h[1] = *(const bf16x8*)(vp + 16);
                acc[f] = wmma_bf16(a.v, bb.v, acc[f]);
            }
        }
        __syncthreads();
    }

    // finalize: rel = acc / l
    float linv[8];
#pragma unroll
    for (int v = 0; v < 8; ++v) linv[v] = 1.0f / run_l[v + 8 * half];
#pragma unroll
    for (int f = 0; f < 4; ++f)
#pragma unroll
        for (int v = 0; v < 8; ++v)
            Ob[(v + 8 * half) * 516 + w * 64 + f * 16 + l16] = acc[f][v] * linv[v];
    __syncthreads();

    // LayerNorm + ReLU: wave w handles rows 2w, 2w+1
    for (int rr = 0; rr < 2; ++rr) {
        int r = w * 2 + rr;
        float s1 = 0.f, s2 = 0.f;
        for (int ci = lane; ci < HH; ci += 32) { float x = Ob[r * 516 + ci]; s1 += x; s2 += x * x; }
        for (int d = 1; d < 32; d <<= 1) { s1 += __shfl_xor(s1, d, 32); s2 += __shfl_xor(s2, d, 32); }
        float mu  = s1 * (1.0f / HH);
        float var = s2 * (1.0f / HH) - mu * mu;
        float inv = rsqrtf(var + 1e-5f);
        for (int ci = lane; ci < HH; ci += 32) {
            float x = Ob[r * 516 + ci];
            float y = (x - mu) * inv * ln_g[ci] + ln_b[ci];
            Ob[r * 516 + ci] = fmaxf(y, 0.f);
        }
    }
    __syncthreads();

    // classifier 512 -> 50
    {
        int r  = tid >> 4;
        int g4 = (tid & 15) * 4;
        size_t grow = (size_t)b * PP + q0 + r;
        for (int oc = g4; oc < g4 + 4 && oc < RR; ++oc) {
            float d = cls_b[oc];
            for (int k = 0; k < HH; ++k) d += Ob[r * 516 + k] * cls_w[(size_t)k * RR + oc];
            Lg[r * 52 + oc] = d;
            out_logits[grow * (RR + 1) + oc] = d;
        }
    }
    __syncthreads();

    if (tid < 16) {
        int r = tid;
        size_t grow = (size_t)b * PP + q0 + r;
        float bin = fus_b[0];
        for (int oc = 0; oc < RR; ++oc) bin += Lg[r * 52 + oc] * fus_w[oc];
        out_logits[grow * (RR + 1) + RR] = bin;
        float score = 1.0f / (1.0f + __expf(-bin));
        int p  = q0 + r;
        int ii = pidx[((size_t)b * PP + p) * 2];
        int jj = pidx[((size_t)b * PP + p) * 2 + 1];
        mats[(size_t)b * NN * NN + ii * NN + jj] = score;
    }
}

// ---------------------------------------------------------------- launcher
extern "C" void kernel_launch(void* const* d_in, const int* in_sizes, int n_in,
                              void* d_out, int out_size, void* d_ws, size_t ws_size,
                              hipStream_t stream) {
    const float* visual_feat = (const float*)d_in[0];
    const float* box_info    = (const float*)d_in[1];
    const float* pred_logits = (const float*)d_in[2];
    const int*   pair_idx    = (const int*)d_in[3];
    const float* obj_sem_emb = (const float*)d_in[4];
    const float* pos_w1 = (const float*)d_in[5];
    const float* pos_b1 = (const float*)d_in[6];
    const float* pos_w2 = (const float*)d_in[7];
    const float* pos_b2 = (const float*)d_in[8];
    const float* geo_w  = (const float*)d_in[9];
    const float* geo_b  = (const float*)d_in[10];
    const float* vis_w  = (const float*)d_in[11];
    const float* vis_b  = (const float*)d_in[12];
    const float* q_w = (const float*)d_in[13];
    const float* q_b = (const float*)d_in[14];
    const float* k_w = (const float*)d_in[15];
    const float* k_b = (const float*)d_in[16];
    const float* v_w = (const float*)d_in[17];
    const float* v_b = (const float*)d_in[18];
    const float* ln_g = (const float*)d_in[19];
    const float* ln_b = (const float*)d_in[20];
    const float* cls_w = (const float*)d_in[21];
    const float* cls_b = (const float*)d_in[22];
    const float* fus_w = (const float*)d_in[23];
    const float* fus_b = (const float*)d_in[24];

    float* out_logits = (float*)d_out;                  // [16128][51]
    float* mats       = out_logits + (size_t)MTOT * (RR + 1);

    uint8_t* ws  = (uint8_t*)d_ws;
    size_t   off = 0;
    auto alloc = [&](size_t bytes) { void* p = ws + off; off = (off + bytes + 255) & ~(size_t)255; return p; };
    __bf16* vis_wt = (__bf16*)alloc((size_t)HH * DIN * 2);
    __bf16* geo_wt = (__bf16*)alloc((size_t)HH * 1792 * 2);
    __bf16* q_wt   = (__bf16*)alloc((size_t)HH * HH * 2);
    __bf16* k_wt   = (__bf16*)alloc((size_t)HH * HH * 2);
    __bf16* v_wt   = (__bf16*)alloc((size_t)HH * HH * 2);
    float*  pos    = (float*)alloc((size_t)BB * NN * GG * 4);
    float*  sem    = (float*)alloc((size_t)BB * NN * EE * 4);
    __bf16* pairb  = (__bf16*)alloc((size_t)MTOT * 1792 * 2);
    __bf16* visH   = (__bf16*)alloc((size_t)MTOT * HH * 2);
    __bf16* geoH   = (__bf16*)alloc((size_t)MTOT * HH * 2);
    __bf16* Qbuf   = (__bf16*)alloc((size_t)MTOT * HH * 2);
    __bf16* Kbuf   = (__bf16*)alloc((size_t)MTOT * HH * 2);
    __bf16* Vtb    = (__bf16*)alloc((size_t)MTOT * HH * 2);  // [b][col][p]

    k_zero<<<64, 256, 0, stream>>>(mats, BB * NN * NN);

    k_cvt_wt<<<(DIN * HH + 255) / 256, 256, 0, stream>>>(vis_w, vis_wt, DIN, HH);
    k_cvt_wt<<<(1792 * HH + 255) / 256, 256, 0, stream>>>(geo_w, geo_wt, 1792, HH);
    k_cvt_wt<<<(HH * HH + 255) / 256, 256, 0, stream>>>(q_w, q_wt, HH, HH);
    k_cvt_wt<<<(HH * HH + 255) / 256, 256, 0, stream>>>(k_w, k_wt, HH, HH);
    k_cvt_wt<<<(HH * HH + 255) / 256, 256, 0, stream>>>(v_w, v_wt, HH, HH);

    k_prep<<<BB * NN, 256, 0, stream>>>(box_info, pred_logits, obj_sem_emb,
                                        pos_w1, pos_b1, pos_w2, pos_b2, pos, sem);
    k_pair<<<MTOT, 256, 0, stream>>>(pair_idx, pos, sem, pairb);

    dim3 gg(MTOT / 32, 1);   // full N=512 per block: A streamed from HBM once
    gemm_bf16<1, false><<<gg, 256, 0, stream>>>(visual_feat, DIN, vis_wt, DIN, vis_b, visH, DIN);
    gemm_bf16<0, false><<<gg, 256, 0, stream>>>(pairb, 1792, geo_wt, 1792, geo_b, geoH, 1792);
    gemm_bf16<0, false><<<gg, 256, 0, stream>>>(visH, HH, q_wt, HH, q_b, Qbuf, HH);
    gemm_bf16<0, false><<<gg, 256, 0, stream>>>(geoH, HH, k_wt, HH, k_b, Kbuf, HH);
    gemm_bf16<0, true ><<<gg, 256, 0, stream>>>(geoH, HH, v_wt, HH, v_b, Vtb, HH);

    k_attn<<<dim3(PP / 16, BB), 256, 0, stream>>>(Qbuf, Kbuf, Vtb, ln_g, ln_b,
                                                  cls_w, cls_b, fus_w, fus_b,
                                                  pair_idx, out_logits, mats);
}
